// GroupedQueryAttention_Qwen_30605936951350
// MI455X (gfx1250) — matrix operand order repaired and checked by reading, not compiled, not run on hardware
//
#include <hip/hip_runtime.h>
#include <hip/hip_bf16.h>

// ---------------- problem constants ----------------
#define B_    2
#define S_    2048
#define DIN_  2048
#define H_    16
#define G_    8
#define DH_   128
#define GS_   (H_ / G_)
#define MROWS_ (B_ * S_)                 // 4096 token rows
#define ATTN_SCALE 0.08838834764831845f  // 1/sqrt(128), folded into Q
#define EPS_  1e-6f

// ---------------- WMMA fragment plumbing (gfx1250, wave32) ----------------
typedef __attribute__((ext_vector_type(16))) __bf16       bf16x16;
typedef __attribute__((ext_vector_type(8)))  float        f32x8;
typedef __attribute__((ext_vector_type(4)))  unsigned int u32x4;

union Frag {
  bf16x16      v;
  unsigned int u[8];
  u32x4        q[2];
  __bf16       e[16];
};

__device__ __forceinline__ f32x8 wmma_bf16(const Frag& a, const Frag& b, f32x8 c) {
  // D(16x16,f32) = A(16x32,bf16) * B(32x16,bf16) + C
  return __builtin_amdgcn_wmma_f32_16x16x32_bf16(
      /*neg_a=*/false, a.v, /*neg_b=*/false, b.v,
      /*c_mod=*/(short)0, c, /*reuse_a=*/false, /*reuse_b=*/false);
}

// A-matrix 16x32 bf16 layout: lane row = lane&15; lanes 0-15 hold K {0..7,16..23},
// lanes 16-31 hold K {8..15,24..31}.  rowp points at (row, kb).
__device__ __forceinline__ Frag load_a32(const __bf16* rowp, int half) {
  Frag f;
  f.q[0] = *(const u32x4*)(rowp + half * 8);        // K = half*8 .. half*8+7
  f.q[1] = *(const u32x4*)(rowp + 16 + half * 8);   // K = 16+half*8 .. +7
  return f;
}

// B-matrix 32x16 bf16 layout: lane col = lane&15; lanes 0-15 hold K {0..7,16..23},
// lanes 16-31 hold K {8..15,24..31} (same as A).  colp points at (k=kb, this col).
__device__ __forceinline__ Frag load_b32(const __bf16* colp, int half) {
  Frag f;
  const __bf16* p = colp + half * 8;
  f.q[0] = *(const u32x4*)(p);
  f.q[1] = *(const u32x4*)(p + 16);
  return f;
}

// ---------------- fp32 -> bf16 convert ----------------
__global__ void cvt_bf16(const float* __restrict__ s, __bf16* __restrict__ d, long n) {
  long i = ((long)blockIdx.x * blockDim.x + threadIdx.x) * 4;
  if (i + 3 < n) {
    float4 v = *(const float4*)(s + i);
    d[i]     = (__bf16)v.x;
    d[i + 1] = (__bf16)v.y;
    d[i + 2] = (__bf16)v.z;
    d[i + 3] = (__bf16)v.w;
  }
}

// ---------------- C[M,N] = A[M,K] * Bw[N,K]^T  (bf16 WMMA, fp32 out) ----------------
// block = 128 threads = 4 waves laid out 2(M) x 2(N); block tile 64 x 128.
// wave tile = 32(M) x 64(N) = 2 A-frags x 4 B-frags -> 8 WMMAs / 32-K step.
// K loop unrolled x2 with ping-pong fragment buffers so next step's
// global_load_b128s are in flight while current step's WMMAs execute.
__global__ __launch_bounds__(128) void gemm_bf16(
    const __bf16* __restrict__ A, const __bf16* __restrict__ Bw,
    float* __restrict__ C, int M, int N, int K)
{
  const int lane = threadIdx.x & 31;
  const int wave = threadIdx.x >> 5;
  const int half = lane >> 4;
  const int ln   = lane & 15;
  const int wm   = wave & 1;
  const int wn   = wave >> 1;
  const int mBase = blockIdx.y * 64 + wm * 32;
  const int nBase = blockIdx.x * 128 + wn * 64;

  f32x8 acc[2][4] = {};

  const __bf16* arow0 = A + (size_t)(mBase + ln) * K;
  const __bf16* arow1 = A + (size_t)(mBase + 16 + ln) * K;
  const __bf16* brow0 = Bw + (size_t)(nBase + 0 * 16 + ln) * K;
  const __bf16* brow1 = Bw + (size_t)(nBase + 1 * 16 + ln) * K;
  const __bf16* brow2 = Bw + (size_t)(nBase + 2 * 16 + ln) * K;
  const __bf16* brow3 = Bw + (size_t)(nBase + 3 * 16 + ln) * K;

  Frag aX[2], bX[4], aY[2], bY[4];

  // prologue: fragments for kb = 0
  aX[0] = load_a32(arow0, half);
  aX[1] = load_a32(arow1, half);
  bX[0] = load_b32(brow0, half);
  bX[1] = load_b32(brow1, half);
  bX[2] = load_b32(brow2, half);
  bX[3] = load_b32(brow3, half);

  for (int kb = 0; kb < K; kb += 64) {
    // issue loads for kb+32 (always valid: K % 64 == 0)
    const int k1 = kb + 32;
    aY[0] = load_a32(arow0 + k1, half);
    aY[1] = load_a32(arow1 + k1, half);
    bY[0] = load_b32(brow0 + k1, half);
    bY[1] = load_b32(brow1 + k1, half);
    bY[2] = load_b32(brow2 + k1, half);
    bY[3] = load_b32(brow3 + k1, half);
    __builtin_prefetch(arow0 + kb + 64, 0, 1);   // global_prefetch next A line

    // compute on kb fragments while kb+32 loads are in flight
#pragma unroll
    for (int mi = 0; mi < 2; ++mi)
#pragma unroll
      for (int t = 0; t < 4; ++t)
        acc[mi][t] = wmma_bf16(aX[mi], bX[t], acc[mi][t]);

    // issue loads for kb+64 while kb+32 WMMAs execute
    const int k2 = kb + 64;
    if (k2 < K) {
      aX[0] = load_a32(arow0 + k2, half);
      aX[1] = load_a32(arow1 + k2, half);
      bX[0] = load_b32(brow0 + k2, half);
      bX[1] = load_b32(brow1 + k2, half);
      bX[2] = load_b32(brow2 + k2, half);
      bX[3] = load_b32(brow3 + k2, half);
    }

#pragma unroll
    for (int mi = 0; mi < 2; ++mi)
#pragma unroll
      for (int t = 0; t < 4; ++t)
        acc[mi][t] = wmma_bf16(aY[mi], bY[t], acc[mi][t]);
  }

  // D layout: col = ln, row = r + 8*half (per 16-row sub-tile)
#pragma unroll
  for (int mi = 0; mi < 2; ++mi) {
#pragma unroll
    for (int t = 0; t < 4; ++t) {
#pragma unroll
      for (int r = 0; r < 8; ++r) {
        size_t m = (size_t)mBase + mi * 16 + r + 8 * half;
        C[m * N + nBase + t * 16 + ln] = acc[mi][t][r];
      }
    }
  }
}

// ---------------- RMSNorm + RoPE + bf16 pack -------------------------------
// One wave per (b,s,slot) row of 128: slot 0..15 = q head, 16..23 = k group,
// 24..31 = v group.  Q gets 1/sqrt(DH) folded in.  V is written TRANSPOSED to
// [B,G,DH,S] so attention PV B-fragments are contiguous dword loads.
__global__ __launch_bounds__(128) void norm_rope_pack(
    const float* __restrict__ qf, const float* __restrict__ kf,
    const float* __restrict__ vf,
    const float* __restrict__ cosT, const float* __restrict__ sinT,
    const float* __restrict__ qscale, const float* __restrict__ kscale,
    __bf16* __restrict__ qb, __bf16* __restrict__ kb, __bf16* __restrict__ vb)
{
  const int lane = threadIdx.x & 31;
  const int wave = threadIdx.x >> 5;
  int rid  = blockIdx.x * 4 + wave;
  int bs   = rid >> 5;
  int slot = rid & 31;
  int b = bs / S_;
  int s = bs - b * S_;

  if (slot >= H_ + G_) {            // ---- V: convert + transpose only
    int g = slot - (H_ + G_);
    const float* src = vf + (size_t)bs * (G_ * DH_) + (size_t)g * DH_;
    __bf16* dst = vb + (((size_t)b * G_ + g) * DH_) * S_;
#pragma unroll
    for (int i = 0; i < 4; ++i) {
      int d = lane + 32 * i;
      dst[(size_t)d * S_ + s] = (__bf16)src[d];
    }
    return;
  }

  bool isq = slot < H_;
  int head = isq ? slot : slot - H_;
  const float* src = isq ? (qf + (size_t)bs * (H_ * DH_) + (size_t)head * DH_)
                         : (kf + (size_t)bs * (G_ * DH_) + (size_t)head * DH_);
  const float* sc = isq ? qscale : kscale;

  float t[4];
  float ss = 0.f;
#pragma unroll
  for (int i = 0; i < 4; ++i) { int d = lane + 32 * i; t[i] = src[d]; ss += t[i] * t[i]; }
#pragma unroll
  for (int o = 16; o >= 1; o >>= 1) ss += __shfl_xor(ss, o, 32);
  float rn = rsqrtf(ss * (1.f / (float)DH_) + EPS_);
#pragma unroll
  for (int i = 0; i < 4; ++i) { int d = lane + 32 * i; t[i] = t[i] * rn * sc[d]; }

  // RoPE: lane holds d, d+32, d+64, d+96 -> both halves of each pair are local.
  float o4[4];
#pragma unroll
  for (int i = 0; i < 2; ++i) {
    int d = lane + 32 * i;                         // d < 64
    float c0 = cosT[(size_t)s * DH_ + d],       s0 = sinT[(size_t)s * DH_ + d];
    float c1 = cosT[(size_t)s * DH_ + d + 64],  s1 = sinT[(size_t)s * DH_ + d + 64];
    o4[i]     = t[i]     * c0 - t[i + 2] * s0;     // x1*cos - x2*sin
    o4[i + 2] = t[i + 2] * c1 + t[i]     * s1;     // x2*cos + x1*sin
  }
  float asc = isq ? ATTN_SCALE : 1.0f;
  __bf16* dst = isq ? (qb + (((size_t)b * H_ + head) * S_ + s) * DH_)
                    : (kb + (((size_t)b * G_ + head) * S_ + s) * DH_);
#pragma unroll
  for (int i = 0; i < 4; ++i) { int d = lane + 32 * i; dst[d] = (__bf16)(o4[i] * asc); }
}

// ---------------- flash attention: one wave per 16-row q-tile ---------------
// Q: [B,H,S,DH] bf16 (pre-scaled), K: [B,G,S,DH] bf16, V: [B,G,DH,S] bf16.
// Per 32-key step: load all 8 K-frags (one clause) -> 8 score WMMAs; issue all
// 8 V-frag loads BEFORE the softmax VALU section (latency hidden under exp/
// shuffle work); causal mask; online softmax; P transposed to A-layout via LDS
// + s_wait_dscnt; 8 PV WMMAs.  Output ctx bf16 in [B*S, H*DH].
__global__ __launch_bounds__(128) void attn_fwd(
    const __bf16* __restrict__ qb, const __bf16* __restrict__ kb,
    const __bf16* __restrict__ vb, __bf16* __restrict__ ctxb)
{
  __shared__ float plds[4][16][36];   // per-wave 16x32 P tile (+bank pad)
  const int lane = threadIdx.x & 31;
  const int wave = threadIdx.x >> 5;
  const int half = lane >> 4;
  const int ln   = lane & 15;

  int gw  = blockIdx.x * 4 + wave;
  int qt  = gw % (S_ / 16);
  int rem = gw / (S_ / 16);
  int h   = rem % H_;
  int b   = rem / H_;
  int grp = h / GS_;
  int q0  = qt * 16;

  const __bf16* Qp = qb + (((size_t)b * H_ + h) * S_ + q0 + ln) * DH_;
  Frag qfr[4];
#pragma unroll
  for (int c = 0; c < 4; ++c) qfr[c] = load_a32(Qp + c * 32, half);

  const __bf16* Kp = kb + (((size_t)b * G_ + grp) * S_) * DH_;
  const __bf16* Vp = vb + (((size_t)b * G_ + grp) * DH_) * S_;

  f32x8 Oc[8] = {};
  float mrow[8], lrow[8];
#pragma unroll
  for (int r = 0; r < 8; ++r) { mrow[r] = -1e30f; lrow[r] = 0.f; }

  const int kend = q0 + 16;                       // causal: keys 0..q0+15
  for (int kbase = 0; kbase < kend; kbase += 32) {
    // ---- load all 8 K fragments up front (clause-able b128 pairs)
    Frag kfr[8];
#pragma unroll
    for (int c = 0; c < 4; ++c) {
      kfr[c]     = load_b32(Kp + (size_t)(kbase + ln) * DH_ + c * 32, half);
      kfr[c + 4] = load_b32(Kp + (size_t)(kbase + 16 + ln) * DH_ + c * 32, half);
    }
    // ---- scores for 32 keys (two 16x16 tiles), K = DH = 128 = 4 x 32
    f32x8 s0 = {}, s1 = {};
#pragma unroll
    for (int c = 0; c < 4; ++c) {
      s0 = wmma_bf16(qfr[c], kfr[c],     s0);
      s1 = wmma_bf16(qfr[c], kfr[c + 4], s1);
    }
    // ---- issue all 8 V fragment loads now; latency hides under softmax VALU
    Frag vfr[8];
#pragma unroll
    for (int c = 0; c < 8; ++c)
      vfr[c] = load_b32(Vp + (size_t)(c * 16 + ln) * S_ + kbase, half);

    // ---- causal mask (only near/diagonal blocks need it)
    if (kbase + 31 > q0) {
#pragma unroll
      for (int r = 0; r < 8; ++r) {
        int qrow = q0 + r + 8 * half;
        if (kbase + ln      > qrow) s0[r] = -1e30f;
        if (kbase + 16 + ln > qrow) s1[r] = -1e30f;
      }
    }
    // ---- online softmax: each D row lives in one VGPR across a 16-lane half
#pragma unroll
    for (int r = 0; r < 8; ++r) {
      float rmax = fmaxf(s0[r], s1[r]);
#pragma unroll
      for (int o = 8; o >= 1; o >>= 1) rmax = fmaxf(rmax, __shfl_xor(rmax, o, 32));
      float mn   = fmaxf(mrow[r], rmax);
      float corr = __expf(mrow[r] - mn);
      mrow[r] = mn;
      float p0 = __expf(s0[r] - mn);
      float p1 = __expf(s1[r] - mn);
      float psum = p0 + p1;
#pragma unroll
      for (int o = 8; o >= 1; o >>= 1) psum += __shfl_xor(psum, o, 32);
      lrow[r] = lrow[r] * corr + psum;
#pragma unroll
      for (int c = 0; c < 8; ++c) Oc[c][r] *= corr;
      plds[wave][r + 8 * half][ln]      = p0;
      plds[wave][r + 8 * half][16 + ln] = p1;
    }
    // wave-local DS ordering: stores above feed cross-lane loads below
    asm volatile("s_wait_dscnt 0x0" ::: "memory");
    // ---- re-load P in A-matrix layout (row = ln), convert to bf16
    Frag pf;
#pragma unroll
    for (int v8 = 0; v8 < 8; ++v8) {
      int koff = (v8 < 4) ? (half * 8 + v8 * 2) : (16 + half * 8 + (v8 - 4) * 2);
      float2 pv = *(const float2*)&plds[wave][ln][koff];
      pf.e[2 * v8]     = (__bf16)pv.x;
      pf.e[2 * v8 + 1] = (__bf16)pv.y;
    }
    // ---- O += P(16x32) x V(32x16 per dh-chunk); V frags already resident
#pragma unroll
    for (int c = 0; c < 8; ++c) Oc[c] = wmma_bf16(pf, vfr[c], Oc[c]);
  }

  float inv[8];
#pragma unroll
  for (int r = 0; r < 8; ++r) inv[r] = 1.f / lrow[r];
#pragma unroll
  for (int c = 0; c < 8; ++c) {
#pragma unroll
    for (int r = 0; r < 8; ++r) {
      size_t row = (size_t)b * S_ + q0 + r + 8 * half;
      ctxb[row * (H_ * DH_) + (size_t)h * DH_ + c * 16 + ln] =
          (__bf16)(Oc[c][r] * inv[r]);
    }
  }
}

// ---------------- host orchestration ----------------
extern "C" void kernel_launch(void* const* d_in, const int* in_sizes, int n_in,
                              void* d_out, int out_size, void* d_ws, size_t ws_size,
                              hipStream_t stream)
{
  const float* x    = (const float*)d_in[0];
  // d_in[1] = boolean mask: causal triu, known statically -> unused
  const float* cosT = (const float*)d_in[2];
  const float* sinT = (const float*)d_in[3];
  const float* Wq   = (const float*)d_in[4];
  const float* Wk   = (const float*)d_in[5];
  const float* Wv   = (const float*)d_in[6];
  const float* Wo   = (const float*)d_in[7];
  const float* qs   = (const float*)d_in[8];
  const float* ks   = (const float*)d_in[9];
  (void)in_sizes; (void)n_in; (void)out_size; (void)ws_size;

  const size_t n_x  = (size_t)MROWS_ * DIN_;        // 8.4M
  const size_t n_wq = (size_t)H_ * DH_ * DIN_;      // 4.2M
  const size_t n_wk = (size_t)G_ * DH_ * DIN_;      // 2.1M
  const size_t n_wo = (size_t)DIN_ * H_ * DH_;      // 4.2M

  char* w = (char*)d_ws;
  auto carve = [&](size_t bytes) -> void* {
    void* p = (void*)w;
    w += (bytes + 255) & ~(size_t)255;
    return p;
  };
  __bf16* xb  = (__bf16*)carve(n_x  * 2);
  __bf16* wqb = (__bf16*)carve(n_wq * 2);
  __bf16* wkb = (__bf16*)carve(n_wk * 2);
  __bf16* wvb = (__bf16*)carve(n_wk * 2);
  __bf16* wob = (__bf16*)carve(n_wo * 2);
  float*  qf  = (float*) carve((size_t)MROWS_ * H_ * DH_ * 4);
  float*  kf  = (float*) carve((size_t)MROWS_ * G_ * DH_ * 4);
  float*  vf  = (float*) carve((size_t)MROWS_ * G_ * DH_ * 4);
  __bf16* qbb = (__bf16*)carve((size_t)MROWS_ * H_ * DH_ * 2);
  __bf16* kbb = (__bf16*)carve((size_t)MROWS_ * G_ * DH_ * 2);
  __bf16* vbb = (__bf16*)carve((size_t)MROWS_ * G_ * DH_ * 2);
  __bf16* ctx = (__bf16*)carve((size_t)MROWS_ * H_ * DH_ * 2);

  // 1) one-time bf16 conversion of activations + weights
  cvt_bf16<<<(unsigned)(n_x  / 4 / 256), 256, 0, stream>>>(x,  xb,  (long)n_x);
  cvt_bf16<<<(unsigned)(n_wq / 4 / 256), 256, 0, stream>>>(Wq, wqb, (long)n_wq);
  cvt_bf16<<<(unsigned)(n_wk / 4 / 256), 256, 0, stream>>>(Wk, wkb, (long)n_wk);
  cvt_bf16<<<(unsigned)(n_wk / 4 / 256), 256, 0, stream>>>(Wv, wvb, (long)n_wk);
  cvt_bf16<<<(unsigned)(n_wo / 4 / 256), 256, 0, stream>>>(Wo, wob, (long)n_wo);

  // 2) Q/K/V projections (WMMA GEMMs), fp32 out for the norm/rope stage
  gemm_bf16<<<dim3((H_ * DH_) / 128, MROWS_ / 64), 128, 0, stream>>>(
      xb, wqb, qf, MROWS_, H_ * DH_, DIN_);
  gemm_bf16<<<dim3((G_ * DH_) / 128, MROWS_ / 64), 128, 0, stream>>>(
      xb, wkb, kf, MROWS_, G_ * DH_, DIN_);
  gemm_bf16<<<dim3((G_ * DH_) / 128, MROWS_ / 64), 128, 0, stream>>>(
      xb, wvb, vf, MROWS_, G_ * DH_, DIN_);

  // 3) RMSNorm + RoPE + bf16 pack (+ V transpose); B*S*(H+2G) rows, 4 waves/blk
  norm_rope_pack<<<(MROWS_ * 32) / 4, 128, 0, stream>>>(
      qf, kf, vf, cosT, sinT, qs, ks, qbb, kbb, vbb);

  // 4) flash attention: B*H*(S/16) q-tiles, 4 waves/block
  attn_fwd<<<(B_ * H_ * (S_ / 16)) / 4, 128, 0, stream>>>(qbb, kbb, vbb, ctx);

  // 5) output projection: out = ctx @ Wo^T (fp32 out to d_out)
  gemm_bf16<<<dim3(DIN_ / 128, MROWS_ / 64), 128, 0, stream>>>(
      ctx, wob, (float*)d_out, MROWS_, DIN_, H_ * DH_);
}
